// ArgumentLogits_87110526697689
// MI455X (gfx1250) — compile-verified, matrix operand order checked by reading, not run on hardware
//
#include <hip/hip_runtime.h>

// ---------------- problem constants (from reference) ----------------
#define BS        256
#define MAX_ARGS  8
#define CTX_DIM   128
#define NODE_DIM  128
#define HIDDEN    512
#define STATE_DIM 512
#define TAC_DIM   128
#define TOTAL_CTX 131072
#define DEF_NUM   20000
#define CVD       256          // CTX_VAL_DIM
#define DIM       257          // CTX_DIM + 1 + NODE_DIM
#define KCAT      (STATE_DIM + TAC_DIM)   // 640
#define LNG_COLS  (MAX_ARGS * DIM)        // 2056
#define MAX_TA    (BS * MAX_ARGS)         // 2048

// ---------------- WMMA types ----------------
typedef __attribute__((ext_vector_type(16))) __bf16 v16bf;
typedef __attribute__((ext_vector_type(8)))  float  v8f;

union BF16x16 { v16bf v; unsigned u[8]; };

// f32 -> bf16 by truncation: one shift (scalar) / one v_perm_b32 (pair).
__device__ __forceinline__ unsigned short f2bf_bits(float x) {
  return (unsigned short)(__float_as_uint(x) >> 16);
}
// pack2(lo,hi) = { bf16(hi), bf16(lo) } in one dword via V_PERM_B32:
// result bytes = hi[3],hi[2],lo[3],lo[2]  -> selector 0x07060302 (S0=hi, S1=lo).
__device__ __forceinline__ unsigned pack2(float lo, float hi) {
  return __builtin_amdgcn_perm(__float_as_uint(hi), __float_as_uint(lo), 0x07060302u);
}

__device__ __forceinline__ v8f wmma_bf16(v16bf a, v16bf b, v8f c) {
  return __builtin_amdgcn_wmma_f32_16x16x32_bf16(false, a, false, b, (short)0, c,
                                                 false, false);
}

// A tile (16 x 32-K chunk) from row-major f32, row = m0 + lane&15.
// VGPR i holds K = (i>>2)*16 + 8*(lane>>4) + (i&3)*2 (+1 in high half).
__device__ __forceinline__ v16bf load_tile_a_f32(const float* A, int ld, int m0,
                                                 int k0, int lane) {
  const float* p = A + (long)(m0 + (lane & 15)) * ld + k0 + ((lane >> 4) << 3);
  BF16x16 r;
#pragma unroll
  for (int i = 0; i < 8; ++i) {
    int k = ((i >> 2) << 4) + ((i & 3) << 1);
    r.u[i] = pack2(p[k], p[k + 1]);
  }
  return r.v;
}

// B tile (32-K chunk x 16) where memory holds B^T as bf16 row-major:
// G[n][k], ld = K-extent. Column n = n0 + lane&15 (clamped to nmax).
// Per VGPR: one contiguous dword = two consecutive K values. Fully coalesced.
__device__ __forceinline__ v16bf load_tile_bT_bf16(const unsigned short* G, int ld,
                                                   int k0, int n0, int lane, int nmax) {
  int n = n0 + (lane & 15); if (n > nmax) n = nmax;
  const unsigned short* p = G + (long)n * ld + k0 + ((lane >> 4) << 3);
  BF16x16 r;
#pragma unroll
  for (int i = 0; i < 8; ++i) {
    int k = ((i >> 2) << 4) + ((i & 3) << 1);
    r.u[i] = *(const unsigned*)(p + k);
  }
  return r.v;
}

// A tile from row-major bf16 (st matrix).
__device__ __forceinline__ v16bf load_tile_a_bf16(const unsigned short* A, int ld,
                                                  int m0, int k0, int lane) {
  const unsigned short* p = A + (long)(m0 + (lane & 15)) * ld + k0 + ((lane >> 4) << 3);
  BF16x16 r;
#pragma unroll
  for (int i = 0; i < 8; ++i) {
    int k = ((i >> 2) << 4) + ((i & 3) << 1);
    r.u[i] = *(const unsigned*)(p + k);
  }
  return r.v;
}

// ================= kernel A: ragged index plumbing (1 block) =================
__global__ void build_indices_kernel(const int* __restrict__ ctx_ids,
                                     const int* __restrict__ arg_cnt,
                                     int* hdr, int* arg_batch_g, int* arg_j_g,
                                     int* pair_start_g, int* ctx_start_g) {
  __shared__ int s_sum[256];
  __shared__ int s_ctx[257];
  __shared__ int s_batch[MAX_TA];
  __shared__ int s_j[MAX_TA];
  __shared__ int s_len[MAX_TA];
  const int t = threadIdx.x;

  // ctx_start[b] = lower_bound(ctx_ids, b)   (ctx_ids sorted)
  { int lo = 0, hi = TOTAL_CTX;
    while (lo < hi) { int mid = (lo + hi) >> 1; if (ctx_ids[mid] < t) lo = mid + 1; else hi = mid; }
    s_ctx[t] = lo;
    if (t == 0) s_ctx[256] = TOTAL_CTX; }

  // exclusive scan of arg_cnt
  const int cnt = arg_cnt[t];
  s_sum[t] = cnt;
  __syncthreads();
  for (int off = 1; off < 256; off <<= 1) {
    int v = (t >= off) ? s_sum[t - off] : 0;
    __syncthreads();
    s_sum[t] += v;
    __syncthreads();
  }
  const int excl = s_sum[t] - cnt;
  const int total_args = s_sum[255];

  for (int a = t; a < MAX_TA; a += 256) { s_batch[a] = 0; s_j[a] = 0; }
  __syncthreads();
  for (int j = 0; j < cnt; ++j) { s_batch[excl + j] = t; s_j[excl + j] = j; }
  __syncthreads();

  // per-arg context lengths
  for (int a = t; a < MAX_TA; a += 256) {
    int v = 0;
    if (a < total_args) { int b = s_batch[a]; v = s_ctx[b + 1] - s_ctx[b]; }
    s_len[a] = v;
  }
  __syncthreads();

  // exclusive scan over 2048 lens: 8-wide serial chunk + block scan of chunk sums
  const int base = t << 3;
  int local[8], csum = 0;
#pragma unroll
  for (int i = 0; i < 8; ++i) { local[i] = csum; csum += s_len[base + i]; }
  s_sum[t] = csum;
  __syncthreads();
  for (int off = 1; off < 256; off <<= 1) {
    int v = (t >= off) ? s_sum[t - off] : 0;
    __syncthreads();
    s_sum[t] += v;
    __syncthreads();
  }
  const int chunk_excl = s_sum[t] - csum;
  const int total_pairs = s_sum[255];
#pragma unroll
  for (int i = 0; i < 8; ++i) pair_start_g[base + i] = chunk_excl + local[i];

  for (int a = t; a < MAX_TA; a += 256) { arg_batch_g[a] = s_batch[a]; arg_j_g[a] = s_j[a]; }
  ctx_start_g[t] = s_ctx[t];
  if (t == 0) { ctx_start_g[256] = TOTAL_CTX; hdr[0] = total_args; hdr[1] = total_pairs; }
}

// ======== weight convert+transpose: out[n*K+k] = bf16(in[k*N+n]) ========
__global__ void convert_transpose_kernel(const float* __restrict__ in,
                                         unsigned short* __restrict__ out,
                                         int K, int N) {
  long total = (long)K * N;
  for (long idx = (long)blockIdx.x * blockDim.x + threadIdx.x; idx < total;
       idx += (long)gridDim.x * blockDim.x) {
    int k = (int)(idx / N), n = (int)(idx % N);
    out[(long)n * K + k] = f2bf_bits(in[idx]);
  }
}

// ======== gk: gather emb_table rows, L2-normalize, store bf16 ========
__global__ void gk_kernel(const float* __restrict__ emb_table,
                          const int* __restrict__ global_context,
                          unsigned short* __restrict__ gk) {
  const int lane = threadIdx.x & 31;
  const int row = blockIdx.x * (blockDim.x >> 5) + (threadIdx.x >> 5);
  if (row >= DEF_NUM) return;
  const float* src = emb_table + (long)global_context[row] * NODE_DIM;
  float4 x = *(const float4*)(src + lane * 4);
  float ss = x.x * x.x + x.y * x.y + x.z * x.z + x.w * x.w;
#pragma unroll
  for (int m = 16; m; m >>= 1) ss += __shfl_xor(ss, m, 32);
  const float inv = 1.0f / (1e-7f + sqrtf(ss));
  unsigned short* dst = gk + (long)row * NODE_DIM + lane * 4;
  *(unsigned*)(dst)     = pack2(x.x * inv, x.y * inv);
  *(unsigned*)(dst + 2) = pack2(x.z * inv, x.w * inv);
}

// ======== st = relu([state|tactic] @ W_st + b_st), bf16 out ========
// 16 M-tiles x 32 N-tiles = 512 tiles; 1 wave/tile.
__global__ void st_gemm_kernel(const float* __restrict__ state_emb,
                               const float* __restrict__ tactic_emb,
                               const unsigned short* __restrict__ WstT,  // 512x640 bf16
                               const float* __restrict__ b_st,
                               unsigned short* __restrict__ st_bf) {
  const int lane = threadIdx.x & 31;
  const int tile = blockIdx.x * (blockDim.x >> 5) + (threadIdx.x >> 5);
  const int m0 = (tile >> 5) << 4, n0 = (tile & 31) << 4;
  v8f acc = {};
#pragma unroll 4
  for (int kc = 0; kc < KCAT / 32; ++kc) {
    const int k0 = kc * 32;
    v16bf a = (k0 < STATE_DIM)
                  ? load_tile_a_f32(state_emb, STATE_DIM, m0, k0, lane)
                  : load_tile_a_f32(tactic_emb, TAC_DIM, m0, k0 - STATE_DIM, lane);
    v16bf b = load_tile_bT_bf16(WstT, KCAT, k0, n0, lane, HIDDEN - 1);
    acc = wmma_bf16(a, b, acc);
  }
  const int col = n0 + (lane & 15);
  const float bias = b_st[col];
#pragma unroll
  for (int v = 0; v < 8; ++v) {
    const int row = m0 + v + ((lane >> 4) << 3);
    float x = acc[v] + bias;
    st_bf[row * HIDDEN + col] = f2bf_bits(x > 0.f ? x : 0.f);
  }
}

// ======== lng = st @ W_q + b_q  (256 x 2056, f32 out) ========
// 16 x 129 tiles (last N tile guarded).
__global__ void lng_gemm_kernel(const unsigned short* __restrict__ st_bf,
                                const unsigned short* __restrict__ WqT,  // 2056x512 bf16
                                const float* __restrict__ b_q,
                                float* __restrict__ lng) {
  const int lane = threadIdx.x & 31;
  const int tile = blockIdx.x * (blockDim.x >> 5) + (threadIdx.x >> 5);
  if (tile >= 16 * 129) return;
  const int m0 = (tile / 129) << 4, n0 = (tile % 129) << 4;
  v8f acc = {};
#pragma unroll 4
  for (int kc = 0; kc < HIDDEN / 32; ++kc) {
    v16bf a = load_tile_a_bf16(st_bf, HIDDEN, m0, kc * 32, lane);
    v16bf b = load_tile_bT_bf16(WqT, HIDDEN, kc * 32, n0, lane, LNG_COLS - 1);
    acc = wmma_bf16(a, b, acc);
  }
  const int col = n0 + (lane & 15);
  if (col < LNG_COLS) {
    const float bias = b_q[col];
#pragma unroll
    for (int v = 0; v < 8; ++v) {
      const int row = m0 + v + ((lane >> 4) << 3);
      lng[(long)row * LNG_COLS + col] = acc[v] + bias;
    }
  }
}

// ======== keys = ctx_vals @ W_key + b_key  (131072 x 128, f32, L2-resident) ====
// One wave per 16-row strip computes all 8 N-tiles (8x A-chunk reuse).
// ctx_vals is the single HBM-resident stream (128 MB) -> prefetch ahead.
__global__ void keys_gemm_kernel(const float* __restrict__ ctx_vals,
                                 const unsigned short* __restrict__ WkT,  // 128x256 bf16
                                 const float* __restrict__ b_key,
                                 float* __restrict__ keys) {
  const int lane = threadIdx.x & 31;
  const int strip = blockIdx.x * (blockDim.x >> 5) + (threadIdx.x >> 5);
  const int m0 = strip << 4;
  const float* arow = ctx_vals + (long)(m0 + (lane & 15)) * CVD + ((lane >> 4) << 3);
  v8f acc[8] = {};
#pragma unroll 2
  for (int kc = 0; kc < CVD / 32; ++kc) {
    if (kc + 2 < CVD / 32)
      __builtin_prefetch(arow + (kc + 2) * 32, 0, 0);   // global_prefetch_b8
    v16bf a = load_tile_a_f32(ctx_vals, CVD, m0, kc * 32, lane);
#pragma unroll
    for (int n = 0; n < 8; ++n) {
      v16bf b = load_tile_bT_bf16(WkT, CVD, kc * 32, n << 4, lane, CTX_DIM - 1);
      acc[n] = wmma_bf16(a, b, acc[n]);
    }
  }
#pragma unroll
  for (int n = 0; n < 8; ++n) {
    const int col = (n << 4) + (lane & 15);
    const float bias = b_key[col];
#pragma unroll
    for (int v = 0; v < 8; ++v) {
      const int row = m0 + v + ((lane >> 4) << 3);
      keys[(long)row * CTX_DIM + col] = acc[n][v] + bias;
    }
  }
}

// ======== none-logits + arange indices ========
__global__ void none_kernel(const float* __restrict__ lng, const int* __restrict__ hdr,
                            const int* __restrict__ arg_batch, const int* __restrict__ arg_j,
                            int* __restrict__ idx_out, float* __restrict__ val_out) {
  const int a = blockIdx.x * blockDim.x + threadIdx.x;
  const int total_args = hdr[0], total_pairs = hdr[1];
  if (a >= total_args) return;
  const long L = (long)total_pairs + total_args + (long)total_args * DEF_NUM;
  idx_out[total_pairs + a] = a;
  val_out[L + total_pairs + a] =
      lng[(long)arg_batch[a] * LNG_COLS + arg_j[a] * DIM + CTX_DIM];
}

// ======== ragged local logits: dot(local_q[a], keys[col]) over arg's ctx run ====
__global__ void local_kernel(const float* __restrict__ lng, const float* __restrict__ keys,
                             const int* __restrict__ hdr, const int* __restrict__ arg_batch,
                             const int* __restrict__ arg_j, const int* __restrict__ pair_start,
                             const int* __restrict__ ctx_start,
                             int* __restrict__ idx_out, float* __restrict__ val_out) {
  const int a = blockIdx.x;
  const int total_args = hdr[0], total_pairs = hdr[1];
  if (a >= total_args) return;
  __shared__ float q[CTX_DIM];
  const int b = arg_batch[a], j = arg_j[a];
  if (threadIdx.x < CTX_DIM)
    q[threadIdx.x] = lng[(long)b * LNG_COLS + j * DIM + threadIdx.x];
  __syncthreads();
  const int cs = ctx_start[b];
  const int len = ctx_start[b + 1] - cs;
  const int p0 = pair_start[a];
  const long L = (long)total_pairs + total_args + (long)total_args * DEF_NUM;
  const int lane = threadIdx.x & 31, wave = threadIdx.x >> 5;
  const float4 qv = *(const float4*)(q + lane * 4);
  for (int i = wave; i < len; i += 8) {
    const float4 kv = *(const float4*)(keys + (long)(cs + i) * CTX_DIM + lane * 4);
    float s = qv.x * kv.x + qv.y * kv.y + qv.z * kv.z + qv.w * kv.w;
#pragma unroll
    for (int m = 16; m; m >>= 1) s += __shfl_xor(s, m, 32);
    if (lane == 0) { idx_out[p0 + i] = a; val_out[L + p0 + i] = s; }
  }
}

// ======== global logits: gq(total_args x 128) @ gk^T (128 x 20000) ========
// 128 x 1250 tiles; streams indices + values straight to d_out.
__global__ void global_gemm_kernel(const float* __restrict__ lng,
                                   const unsigned short* __restrict__ gk,
                                   const int* __restrict__ hdr,
                                   const int* __restrict__ arg_batch,
                                   const int* __restrict__ arg_j,
                                   int* __restrict__ idx_out, float* __restrict__ val_out) {
  const int lane = threadIdx.x & 31;
  const int tile = blockIdx.x * (blockDim.x >> 5) + (threadIdx.x >> 5);
  const int mt = tile / 1250, nt = tile % 1250;
  const int total_args = hdr[0], total_pairs = hdr[1];
  const int m0 = mt << 4;
  if (m0 >= total_args) return;

  // gathered A rows: lng[arg_batch[a]][arg_j[a]*257 + 129 .. +256]
  int a = m0 + (lane & 15);
  const int ac = (a < total_args) ? a : (total_args - 1);
  const float* arow = lng + (long)arg_batch[ac] * LNG_COLS + arg_j[ac] * DIM + CTX_DIM + 1;
  const int kh = (lane >> 4) << 3;

  v8f acc = {};
#pragma unroll
  for (int kc = 0; kc < NODE_DIM / 32; ++kc) {
    BF16x16 ra;
    const float* p = arow + kc * 32 + kh;
#pragma unroll
    for (int i = 0; i < 8; ++i) {
      int k = ((i >> 2) << 4) + ((i & 3) << 1);
      ra.u[i] = pack2(p[k], p[k + 1]);
    }
    v16bf b = load_tile_bT_bf16(gk, NODE_DIM, kc * 32, nt << 4, lane, DEF_NUM - 1);
    acc = wmma_bf16(ra.v, b, acc);
  }

  const long base = (long)total_pairs + total_args;          // start of repeat-section
  const long L = base + (long)total_args * DEF_NUM;          // start of values in d_out
  const int col = (nt << 4) + (lane & 15);
#pragma unroll
  for (int v = 0; v < 8; ++v) {
    const int row = m0 + v + ((lane >> 4) << 3);
    if (row < total_args) {
      const long o = base + (long)row * DEF_NUM + col;
      idx_out[o] = row;
      val_out[L + o] = acc[v];
    }
  }
}

// =============================== launcher ===============================
extern "C" void kernel_launch(void* const* d_in, const int* in_sizes, int n_in,
                              void* d_out, int out_size, void* d_ws, size_t ws_size,
                              hipStream_t stream) {
  const float* ctx_vals       = (const float*)d_in[0];
  const float* state_emb      = (const float*)d_in[1];
  const float* tactic_emb     = (const float*)d_in[2];
  const float* emb_table      = (const float*)d_in[3];
  const float* W_key          = (const float*)d_in[4];
  const float* b_key          = (const float*)d_in[5];
  const float* W_st           = (const float*)d_in[6];
  const float* b_st           = (const float*)d_in[7];
  const float* W_q            = (const float*)d_in[8];
  const float* b_q            = (const float*)d_in[9];
  const int*   ctx_ids        = (const int*)d_in[10];
  const int*   arg_cnt        = (const int*)d_in[11];
  const int*   global_context = (const int*)d_in[12];

  // ---- workspace carve (16B-aligned sections; ~74 MB total) ----
  char* ws = (char*)d_ws;
  int* hdr              = (int*)ws;            ws += 256;
  int* ctx_start        = (int*)ws;            ws += 288 * sizeof(int);
  int* arg_batch        = (int*)ws;            ws += MAX_TA * sizeof(int);
  int* arg_j            = (int*)ws;            ws += MAX_TA * sizeof(int);
  int* pair_start       = (int*)ws;            ws += MAX_TA * sizeof(int);
  unsigned short* WstT  = (unsigned short*)ws; ws += (size_t)HIDDEN * KCAT * 2;
  unsigned short* WqT   = (unsigned short*)ws; ws += (size_t)LNG_COLS * HIDDEN * 2;
  unsigned short* WkT   = (unsigned short*)ws; ws += (size_t)CTX_DIM * CVD * 2;
  unsigned short* st_bf = (unsigned short*)ws; ws += (size_t)BS * HIDDEN * 2;
  float* lng            = (float*)ws;          ws += (size_t)BS * LNG_COLS * 4;
  unsigned short* gk    = (unsigned short*)ws; ws += (size_t)DEF_NUM * NODE_DIM * 2;
  float* keys           = (float*)ws;          ws += (size_t)TOTAL_CTX * CTX_DIM * 4;

  int*   idx_out = (int*)d_out;
  float* val_out = (float*)d_out;

  build_indices_kernel<<<1, 256, 0, stream>>>(ctx_ids, arg_cnt, hdr, arg_batch,
                                              arg_j, pair_start, ctx_start);
  convert_transpose_kernel<<<256, 256, 0, stream>>>(W_st, WstT, KCAT, HIDDEN);
  convert_transpose_kernel<<<512, 256, 0, stream>>>(W_q, WqT, HIDDEN, LNG_COLS);
  convert_transpose_kernel<<<64, 256, 0, stream>>>(W_key, WkT, CVD, CTX_DIM);
  gk_kernel<<<DEF_NUM / 8, 256, 0, stream>>>(emb_table, global_context, gk);

  st_gemm_kernel<<<(16 * 32) / 8, 256, 0, stream>>>(state_emb, tactic_emb, WstT, b_st, st_bf);
  lng_gemm_kernel<<<(16 * 129 + 7) / 8, 256, 0, stream>>>(st_bf, WqT, b_q, lng);
  keys_gemm_kernel<<<(TOTAL_CTX / 16) / 8, 256, 0, stream>>>(ctx_vals, WkT, b_key, keys);

  none_kernel<<<MAX_TA / 256, 256, 0, stream>>>(lng, hdr, arg_batch, arg_j, idx_out, val_out);
  local_kernel<<<MAX_TA, 256, 0, stream>>>(lng, keys, hdr, arg_batch, arg_j,
                                           pair_start, ctx_start, idx_out, val_out);
  global_gemm_kernel<<<(128 * 1250) / 8, 256, 0, stream>>>(lng, gk, hdr, arg_batch,
                                                           arg_j, idx_out, val_out);
}
